// PointNetTransformerBackbone_19121194402085
// MI455X (gfx1250) — matrix operand order repaired and checked by reading, not compiled, not run on hardware
//
#include <hip/hip_runtime.h>
#include <cstdint>
#include <cstddef>

// ---------------------------------------------------------------------------
// Types / constants
// ---------------------------------------------------------------------------
typedef _Float16 h16;
typedef __attribute__((ext_vector_type(16))) _Float16 v16h;
typedef __attribute__((ext_vector_type(8)))  float    v8f;

constexpr int   N_PTS   = 8192;
constexpr int   DMODEL  = 256;
constexpr int   KNN     = 16;
constexpr int   NHEAD   = 8;
constexpr int   DH      = 32;          // head dim == wave32
constexpr int   M_ANCH  = 819;         // int(8192*0.1)
constexpr int   M_PAD   = 832;         // round up to 16
constexpr int   STRIDE  = 10;          // 8192 // 819
constexpr float LN_EPS  = 1e-5f;
constexpr float ATT_SCALE = 0.17677669529663687f; // 32^-0.5

__device__ __forceinline__ float gelu_f(float x) {
  return 0.5f * x * (1.0f + erff(x * 0.7071067811865476f));
}
__device__ __forceinline__ float wave_sum(float v) {
#pragma unroll
  for (int off = 16; off > 0; off >>= 1) v += __shfl_xor(v, off, 32);
  return v;
}

// ---------------------------------------------------------------------------
// WMMA GEMM:  C[M x Nout] = A_f16[M x K] * W^T_f16[Nout x K] + bias
// One wave computes a 16x64 tile: one A fragment (two contiguous b128 loads
// per lane, per the 16-bit A 16x32 layout) is reused across 4 B fragments /
// 4 v8f accumulators -> 4x A-load amortization, 4 WMMAs per K-step.
// A pad rows (m >= M_actual) are loaded unguarded: they only affect C rows
// that the store guard drops, and every A buffer is allocated Mpad rows.
// K is a template parameter (64/256/512/1024) for full unroll + immediate
// offsets -> pure global_load_b128, no flat path, no DScnt coupling.
// ---------------------------------------------------------------------------
union FragU { v16h v; uint4 u[2]; };

template <int K>
__global__ __launch_bounds__(256) void wmma_gemm_kernel(
    const h16* __restrict__ A, int lda, int M_actual, int Mpad,
    const h16* __restrict__ Wt, int Nout,
    const float* __restrict__ bias,
    float* __restrict__ o32, int ld32,
    h16* __restrict__ o16, int ld16,
    int do_gelu) {
  const int wave  = threadIdx.x >> 5;
  const int lane  = threadIdx.x & 31;
  const int half  = lane >> 4;
  const int l16   = lane & 15;
  const int tiles_m = Mpad >> 4;
  const int tiles_n = Nout >> 6;      // 64-wide N tiles
  const int wid = blockIdx.x * 8 + wave;
  if (wid >= tiles_m * tiles_n) return;
  const int tm = wid % tiles_m;
  const int tn = wid / tiles_m;

  const int arow = tm * 16 + l16;
  const h16* __restrict__ Arow = A + (size_t)arow * lda;

  const h16* Wrow[4];
  int   ncol[4];
  v8f   c[4];
#pragma unroll
  for (int j = 0; j < 4; ++j) {
    const int wr = tn * 64 + j * 16 + l16;
    Wrow[j] = Wt + (size_t)wr * K;
    ncol[j] = wr;
    const float bv = bias ? bias[wr] : 0.0f;
#pragma unroll
    for (int r = 0; r < 8; ++r) c[j][r] = bv;
  }

#pragma unroll
  for (int kc = 0; kc < (K >> 5); ++kc) {
    const int kb = kc * 32;
    FragU a;
    a.u[0] = *(const uint4*)(Arow + kb + half * 8);
    a.u[1] = *(const uint4*)(Arow + kb + 16 + half * 8);
#pragma unroll
    for (int j = 0; j < 4; ++j) {
      FragU b;
      b.u[0] = *(const uint4*)(Wrow[j] + kb + half * 8);
      b.u[1] = *(const uint4*)(Wrow[j] + kb + 16 + half * 8);
      c[j] = __builtin_amdgcn_wmma_f32_16x16x32_f16(
          false, a.v, false, b.v, (short)0, c[j], false, false);
    }
  }

#pragma unroll
  for (int r = 0; r < 8; ++r) {
    const int m = tm * 16 + r + half * 8;
    if (m >= M_actual) continue;
#pragma unroll
    for (int j = 0; j < 4; ++j) {
      float val = c[j][r];
      if (do_gelu) val = gelu_f(val);
      if (o32) o32[(size_t)m * ld32 + ncol[j]] = val;
      if (o16) o16[(size_t)m * ld16 + ncol[j]] = (h16)val;
    }
  }
}

// ---------------------------------------------------------------------------
// kNN: top-16 nearest (including self) over all 8192 points.
// pos staged in LDS (96KB). One wave per query (4 queries/wave). Per-lane
// sorted top-16 over N/32 candidates, then 16-round shuffle-butterfly merge.
// ---------------------------------------------------------------------------
__global__ __launch_bounds__(256) void knn_kernel(
    const float* __restrict__ pos, int* __restrict__ idx_out) {
  extern __shared__ float sm[];
  float* sx = sm;
  float* sy = sm + N_PTS;
  float* sz = sm + 2 * N_PTS;
  float* md = sm + 3 * N_PTS;              // 8 waves * 512 floats
  int*   mi = (int*)(sm + 3 * N_PTS + 8 * 512);

  for (int t = threadIdx.x; t < N_PTS; t += 256) {
    sx[t] = pos[3 * t + 0];
    sy[t] = pos[3 * t + 1];
    sz[t] = pos[3 * t + 2];
  }
  __syncthreads();

  const int wave = threadIdx.x >> 5;
  const int lane = threadIdx.x & 31;
  const int mbase = wave * 512;

  for (int qi = 0; qi < 4; ++qi) {
    const int q = blockIdx.x * 32 + wave * 4 + qi;
    const float qx = sx[q], qy = sy[q], qz = sz[q];

    float best[KNN];
    int   bidx[KNN];
#pragma unroll
    for (int t = 0; t < KNN; ++t) { best[t] = 3.4e38f; bidx[t] = -1; }

    for (int j = lane; j < N_PTS; j += 32) {
      const float dx = sx[j] - qx, dy = sy[j] - qy, dz = sz[j] - qz;
      const float d2 = dx * dx + dy * dy + dz * dz;
      if (d2 < best[KNN - 1]) {
        best[KNN - 1] = d2; bidx[KNN - 1] = j;
#pragma unroll
        for (int t = KNN - 1; t > 0; --t) {
          if (best[t] < best[t - 1]) {
            float tf = best[t]; best[t] = best[t - 1]; best[t - 1] = tf;
            int   ti = bidx[t]; bidx[t] = bidx[t - 1]; bidx[t - 1] = ti;
          }
        }
      }
    }
#pragma unroll
    for (int t = 0; t < KNN; ++t) {
      md[mbase + lane * KNN + t] = best[t];
      mi[mbase + lane * KNN + t] = bidx[t];
    }
    __syncthreads();

    int p = 0;
    for (int s = 0; s < KNN; ++s) {
      float v  = (p < KNN) ? md[mbase + lane * KNN + p] : 3.4e38f;
      int   vi = (p < KNN) ? mi[mbase + lane * KNN + p] : -1;
      int   vl = lane;
#pragma unroll
      for (int off = 16; off > 0; off >>= 1) {
        float ov = __shfl_xor(v, off, 32);
        int   oi = __shfl_xor(vi, off, 32);
        int   ol = __shfl_xor(vl, off, 32);
        if (ov < v || (ov == v && ol < vl)) { v = ov; vi = oi; vl = ol; }
      }
      if (lane == vl) ++p;
      if (lane == 0) idx_out[q * KNN + s] = vi;
    }
    __syncthreads();
  }
}

// ---------------------------------------------------------------------------
// Embed stage 1: h = gelu(ln(in3 @ W1 + b1))  -> f16 (rows x 64)
// One wave per row; lane owns features {lane, lane+32}.
// ---------------------------------------------------------------------------
__global__ __launch_bounds__(256) void embed1_kernel(
    const float* __restrict__ in3, const float* __restrict__ W1,
    const float* __restrict__ b1, const float* __restrict__ g,
    const float* __restrict__ bt, h16* __restrict__ out) {
  const int wave = threadIdx.x >> 5;
  const int lane = threadIdx.x & 31;
  const int row = blockIdx.x * 8 + wave;
  if (row >= N_PTS) return;
  const float c0 = in3[row * 3 + 0], c1 = in3[row * 3 + 1], c2 = in3[row * 3 + 2];
  const int f0 = lane, f1 = lane + 32;
  float o0 = c0 * W1[f0] + c1 * W1[64 + f0] + c2 * W1[128 + f0] + b1[f0];
  float o1 = c0 * W1[f1] + c1 * W1[64 + f1] + c2 * W1[128 + f1] + b1[f1];
  const float mean = wave_sum(o0 + o1) * (1.0f / 64.0f);
  const float d0 = o0 - mean, d1 = o1 - mean;
  const float var = wave_sum(d0 * d0 + d1 * d1) * (1.0f / 64.0f);
  const float rstd = rsqrtf(var + LN_EPS);
  out[row * 64 + f0] = (h16)gelu_f(d0 * rstd * g[f0] + bt[f0]);
  out[row * 64 + f1] = (h16)gelu_f(d1 * rstd * g[f1] + bt[f1]);
}

// ---------------------------------------------------------------------------
// LayerNorm over D=256 (optional residual, optional GELU, dual f32/f16 out).
// One wave per row; lane owns d = lane + 32*i.
// ---------------------------------------------------------------------------
__global__ __launch_bounds__(256) void ln_kernel(
    const float* __restrict__ in, int ld_in,
    const float* __restrict__ res, int ld_res,
    const float* __restrict__ g, const float* __restrict__ b,
    int rows, int do_gelu,
    float* __restrict__ o32, int ld32,
    h16* __restrict__ o16, int ld16) {
  const int wave = threadIdx.x >> 5;
  const int lane = threadIdx.x & 31;
  const int row = blockIdx.x * 8 + wave;
  if (row >= rows) return;
  float vals[8];
  float s = 0.0f;
#pragma unroll
  for (int i = 0; i < 8; ++i) {
    const int d = lane + i * 32;
    float t = in[(size_t)row * ld_in + d];
    if (res) t += res[(size_t)row * ld_res + d];
    vals[i] = t;
    s += t;
  }
  const float mean = wave_sum(s) * (1.0f / 256.0f);
  float sq = 0.0f;
#pragma unroll
  for (int i = 0; i < 8; ++i) { const float d = vals[i] - mean; sq += d * d; }
  const float rstd = rsqrtf(wave_sum(sq) * (1.0f / 256.0f) + LN_EPS);
#pragma unroll
  for (int i = 0; i < 8; ++i) {
    const int d = lane + i * 32;
    float y = (vals[i] - mean) * rstd * g[d] + b[d];
    if (do_gelu) y = gelu_f(y);
    if (o32) o32[(size_t)row * ld32 + d] = y;
    if (o16) o16[(size_t)row * ld16 + d] = (h16)y;
  }
}

// ---------------------------------------------------------------------------
// Relative-position bias MLP: bias[n,k,h] = (gelu(rel@pw1+pb1))@pw2 + pb2
// One thread per (n,k) pair; streams the 64 hidden units.
// ---------------------------------------------------------------------------
__global__ __launch_bounds__(256) void relbias_kernel(
    const float* __restrict__ pos, const int* __restrict__ idx,
    const float* __restrict__ pw1, const float* __restrict__ pb1,
    const float* __restrict__ pw2, const float* __restrict__ pb2,
    float* __restrict__ bias_out) {
  const int t = blockIdx.x * 256 + threadIdx.x;
  if (t >= N_PTS * KNN) return;
  const int n = t / KNN;
  const int j = idx[t];
  const float rx = pos[3 * j + 0] - pos[3 * n + 0];
  const float ry = pos[3 * j + 1] - pos[3 * n + 1];
  const float rz = pos[3 * j + 2] - pos[3 * n + 2];
  float acc[NHEAD];
#pragma unroll
  for (int o = 0; o < NHEAD; ++o) acc[o] = pb2[o];
  for (int jj = 0; jj < 64; ++jj) {
    const float hv = gelu_f(rx * pw1[jj] + ry * pw1[64 + jj] + rz * pw1[128 + jj] + pb1[jj]);
#pragma unroll
    for (int o = 0; o < NHEAD; ++o) acc[o] += hv * pw2[jj * NHEAD + o];
  }
#pragma unroll
  for (int o = 0; o < NHEAD; ++o) bias_out[t * NHEAD + o] = acc[o];
}

// ---------------------------------------------------------------------------
// Local attention (k=16 neighbors, dh=32 == one wave lane per channel).
// ---------------------------------------------------------------------------
__global__ __launch_bounds__(256) void local_attn_kernel(
    const float* __restrict__ q, const float* __restrict__ kb,
    const float* __restrict__ vb, const int* __restrict__ idx,
    const float* __restrict__ bias, h16* __restrict__ out16) {
  const int wave = threadIdx.x >> 5;
  const int lane = threadIdx.x & 31;
  const int wid = blockIdx.x * 8 + wave;
  if (wid >= N_PTS * NHEAD) return;
  const int n = wid >> 3;
  const int h = wid & 7;
  const float qv = q[(size_t)n * DMODEL + h * DH + lane];
  float av[KNN];
  int   ji[KNN];
#pragma unroll
  for (int kk = 0; kk < KNN; ++kk) {
    const int j = idx[n * KNN + kk];
    ji[kk] = j;
    const float kv = kb[(size_t)j * DMODEL + h * DH + lane];
    av[kk] = wave_sum(qv * kv) * ATT_SCALE + bias[(n * KNN + kk) * NHEAD + h];
  }
  float mx = -3.4e38f;
#pragma unroll
  for (int kk = 0; kk < KNN; ++kk) mx = fmaxf(mx, av[kk]);
  float sm = 0.0f;
#pragma unroll
  for (int kk = 0; kk < KNN; ++kk) { av[kk] = expf(av[kk] - mx); sm += av[kk]; }
  float outd = 0.0f;
#pragma unroll
  for (int kk = 0; kk < KNN; ++kk)
    outd += av[kk] * vb[(size_t)ji[kk] * DMODEL + h * DH + lane];
  out16[(size_t)n * DMODEL + h * DH + lane] = (h16)(outd / sm);
}

// ---------------------------------------------------------------------------
// Global attention over M=819 anchors, flash-style single pass.
// qkv laid out (M x 768): q|k|v slots of 256 each.
// ---------------------------------------------------------------------------
__global__ __launch_bounds__(256) void global_attn_kernel(
    const float* __restrict__ qkv, h16* __restrict__ og) {
  const int wave = threadIdx.x >> 5;
  const int lane = threadIdx.x & 31;
  const int wid = blockIdx.x * 8 + wave;
  if (wid >= M_ANCH * NHEAD) return;
  const int n = wid >> 3;
  const int h = wid & 7;
  const float qd = qkv[(size_t)n * 768 + h * DH + lane];
  float mx = -3.4e38f, sm = 0.0f, acc = 0.0f;
  for (int m = 0; m < M_ANCH; ++m) {
    const float kd = qkv[(size_t)m * 768 + 256 + h * DH + lane];
    const float s = wave_sum(qd * kd) * ATT_SCALE;
    const float nm = fmaxf(mx, s);
    const float corr = expf(mx - nm);
    const float e = expf(s - nm);
    sm = sm * corr + e;
    acc = acc * corr + e * qkv[(size_t)m * 768 + 512 + h * DH + lane];
    mx = nm;
  }
  og[(size_t)n * DMODEL + h * DH + lane] = (h16)(acc / sm);
}

// ---------------------------------------------------------------------------
// Small movers
// ---------------------------------------------------------------------------
__global__ __launch_bounds__(256) void transpose_w_kernel(
    const float* __restrict__ W, h16* __restrict__ Wt, int Din, int Dout) {
  const int t = blockIdx.x * 256 + threadIdx.x;
  if (t >= Din * Dout) return;
  const int kk = t / Dout;
  const int n = t % Dout;
  Wt[(size_t)n * Din + kk] = (h16)W[t];
}

__global__ __launch_bounds__(256) void gather_anchor_kernel(
    const float* __restrict__ xl, h16* __restrict__ xg16, float* __restrict__ xg32) {
  const int m = blockIdx.x;          // 0..818
  const int c = threadIdx.x;
  const float v = xl[(size_t)(m * STRIDE) * DMODEL + c];
  xg16[(size_t)m * DMODEL + c] = (h16)v;
  xg32[(size_t)m * DMODEL + c] = v;
}

__global__ __launch_bounds__(256) void upsample_kernel(
    const float* __restrict__ xg2, h16* __restrict__ cat1, h16* __restrict__ cat2) {
  const int n = blockIdx.x;          // 0..8191
  const int c = threadIdx.x;
  int g = n / STRIDE; if (g > M_ANCH - 1) g = M_ANCH - 1;
  const float v = xg2[(size_t)g * DMODEL + c];
  cat1[(size_t)n * 512 + 256 + c] = (h16)v;   // [x_local | xg_up]
  cat2[(size_t)n * 512 + c]       = (h16)v;   // [xg_up | x_fused]
}

// ---------------------------------------------------------------------------
// Host driver
// ---------------------------------------------------------------------------
extern "C" void kernel_launch(void* const* d_in, const int* in_sizes, int n_in,
                              void* d_out, int out_size, void* d_ws, size_t ws_size,
                              hipStream_t stream) {
  (void)in_sizes; (void)n_in; (void)out_size; (void)ws_size;
  const float* pos  = (const float*)d_in[0];
  const float* feat = (const float*)d_in[1];
  const float* P[54];
  for (int i = 0; i < 54; ++i) P[i] = (const float*)d_in[2 + i];
  float* out = (float*)d_out;

  uint8_t* ws = (uint8_t*)d_ws;
  size_t off = 0;
  auto alloc = [&](size_t bytes) -> uint8_t* {
    off = (off + 255) & ~size_t(255);
    uint8_t* p = ws + off;
    off += bytes;
    return p;
  };

  // scratch layout
  int*   idxb   = (int*)  alloc((size_t)N_PTS * KNN * 4);
  h16* cw2t   = (h16*)alloc(256 * 64 * 2);
  h16* fw2t   = (h16*)alloc(256 * 64 * 2);
  h16* efuwt  = (h16*)alloc(256 * 512 * 2);
  h16* wqt    = (h16*)alloc(256 * 256 * 2);
  h16* wkt    = (h16*)alloc(256 * 256 * 2);
  h16* wvt    = (h16*)alloc(256 * 256 * 2);
  h16* wot    = (h16*)alloc(256 * 256 * 2);
  h16* wqkvt  = (h16*)alloc(768 * 256 * 2);
  h16* gawot  = (h16*)alloc(256 * 256 * 2);
  h16* gafw1t = (h16*)alloc(1024 * 256 * 2);
  h16* gafw2t = (h16*)alloc(256 * 1024 * 2);
  h16* fuwt   = (h16*)alloc(256 * 512 * 2);
  h16* fpw1t  = (h16*)alloc(256 * 512 * 2);
  h16* fpw2t  = (h16*)alloc(256 * 256 * 2);
  h16* hce    = (h16*)alloc((size_t)N_PTS * 64 * 2);
  h16* hfe    = (h16*)alloc((size_t)N_PTS * 64 * 2);
  h16* xcat   = (h16*)alloc((size_t)N_PTS * 512 * 2);
  float* sA   = (float*)alloc((size_t)N_PTS * DMODEL * 4);
  float* x32  = (float*)alloc((size_t)N_PTS * DMODEL * 4);
  h16* x16    = (h16*)alloc((size_t)N_PTS * DMODEL * 2);
  float* qb   = (float*)alloc((size_t)N_PTS * DMODEL * 4);
  float* kbf  = (float*)alloc((size_t)N_PTS * DMODEL * 4);
  float* vbf  = (float*)alloc((size_t)N_PTS * DMODEL * 4);
  float* biasb= (float*)alloc((size_t)N_PTS * KNN * NHEAD * 4);
  h16* aout16 = (h16*)alloc((size_t)N_PTS * DMODEL * 2);
  float* xl32 = (float*)alloc((size_t)N_PTS * DMODEL * 4);
  h16* cat1   = (h16*)alloc((size_t)N_PTS * 512 * 2);
  h16* cat2   = (h16*)alloc((size_t)N_PTS * 512 * 2);
  h16* xg16   = (h16*)alloc((size_t)M_PAD * DMODEL * 2);
  float* xg0  = (float*)alloc((size_t)M_PAD * DMODEL * 4);
  float* qkvb = (float*)alloc((size_t)M_PAD * 768 * 4);
  h16* og16   = (h16*)alloc((size_t)M_PAD * DMODEL * 2);
  float* sGA  = (float*)alloc((size_t)M_PAD * DMODEL * 4);
  float* xg1f = (float*)alloc((size_t)M_PAD * DMODEL * 4);
  h16* xg1h   = (h16*)alloc((size_t)M_PAD * DMODEL * 2);
  h16* h1g    = (h16*)alloc((size_t)M_PAD * 1024 * 2);
  float* xg2f = (float*)alloc((size_t)M_PAD * DMODEL * 4);
  h16* h1f    = (h16*)alloc((size_t)N_PTS * DMODEL * 2);

  auto tr = [&](const float* W, h16* Wt, int Din, int Dout) {
    int total = Din * Dout;
    transpose_w_kernel<<<(total + 255) / 256, 256, 0, stream>>>(W, Wt, Din, Dout);
  };
  auto gemm = [&](const h16* A, int lda, int Mact, int Mpad,
                  const h16* Wt, int K, int Nout, const float* bias,
                  float* o32, int l32, h16* o16, int l16, int gel) {
    int tiles = (Mpad / 16) * (Nout / 64);
    int blocks = (tiles + 7) / 8;
    switch (K) {
      case 64:
        wmma_gemm_kernel<64><<<blocks, 256, 0, stream>>>(
            A, lda, Mact, Mpad, Wt, Nout, bias, o32, l32, o16, l16, gel);
        break;
      case 256:
        wmma_gemm_kernel<256><<<blocks, 256, 0, stream>>>(
            A, lda, Mact, Mpad, Wt, Nout, bias, o32, l32, o16, l16, gel);
        break;
      case 512:
        wmma_gemm_kernel<512><<<blocks, 256, 0, stream>>>(
            A, lda, Mact, Mpad, Wt, Nout, bias, o32, l32, o16, l16, gel);
        break;
      case 1024:
        wmma_gemm_kernel<1024><<<blocks, 256, 0, stream>>>(
            A, lda, Mact, Mpad, Wt, Nout, bias, o32, l32, o16, l16, gel);
        break;
      default:
        break;
    }
  };
  auto lnk = [&](const float* in, int ldin, const float* res, int ldres,
                 const float* g, const float* b, int rows, int gel,
                 float* o32, int l32, h16* o16, int l16) {
    ln_kernel<<<(rows + 7) / 8, 256, 0, stream>>>(
        in, ldin, res, ldres, g, b, rows, gel, o32, l32, o16, l16);
  };

  // --- weight transposition to f16 (row = output channel, contiguous K) ---
  tr(P[4],  cw2t,   64,  256);
  tr(P[10], fw2t,   64,  256);
  tr(P[12], efuwt,  512, 256);
  tr(P[16], wqt,    256, 256);
  tr(P[18], wkt,    256, 256);
  tr(P[20], wvt,    256, 256);
  tr(P[22], wot,    256, 256);
  tr(P[30], wqkvt,  256, 768);
  tr(P[32], gawot,  256, 256);
  tr(P[38], gafw1t, 256, 1024);
  tr(P[40], gafw2t, 1024, 256);
  tr(P[42], fuwt,   512, 256);
  tr(P[46], fpw1t,  512, 256);
  tr(P[50], fpw2t,  256, 256);

  // --- kNN (96KB pos + 32KB merge scratch in LDS) ---
  knn_kernel<<<256, 256, (3 * N_PTS + 8 * 512 + 8 * 512) * 4, stream>>>(pos, idxb);

  // --- embeddings ---
  embed1_kernel<<<N_PTS / 8, 256, 0, stream>>>(pos,  P[0], P[1], P[2], P[3], hce);
  embed1_kernel<<<N_PTS / 8, 256, 0, stream>>>(feat, P[6], P[7], P[8], P[9], hfe);
  gemm(hce, 64, N_PTS, N_PTS, cw2t, 64, 256, P[5],  nullptr, 0, xcat,       512, 0);
  gemm(hfe, 64, N_PTS, N_PTS, fw2t, 64, 256, P[11], nullptr, 0, xcat + 256, 512, 0);
  gemm(xcat, 512, N_PTS, N_PTS, efuwt, 512, 256, P[13], sA, 256, nullptr, 0, 0);
  lnk(sA, 256, nullptr, 0, P[14], P[15], N_PTS, 1, x32, 256, x16, 256);

  // --- local attention ---
  gemm(x16, 256, N_PTS, N_PTS, wqt, 256, 256, P[17], qb,  256, nullptr, 0, 0);
  gemm(x16, 256, N_PTS, N_PTS, wkt, 256, 256, P[19], kbf, 256, nullptr, 0, 0);
  gemm(x16, 256, N_PTS, N_PTS, wvt, 256, 256, P[21], vbf, 256, nullptr, 0, 0);
  relbias_kernel<<<(N_PTS * KNN + 255) / 256, 256, 0, stream>>>(
      pos, idxb, P[24], P[25], P[26], P[27], biasb);
  local_attn_kernel<<<N_PTS * NHEAD / 8, 256, 0, stream>>>(
      qb, kbf, vbf, idxb, biasb, aout16);
  gemm(aout16, 256, N_PTS, N_PTS, wot, 256, 256, P[23], sA, 256, nullptr, 0, 0);
  lnk(sA, 256, x32, 256, P[28], P[29], N_PTS, 0, xl32, 256, cat1, 512);

  // --- global branch (M=819 anchors, padded to 832) ---
  gather_anchor_kernel<<<M_ANCH, 256, 0, stream>>>(xl32, xg16, xg0);
  gemm(xg16, 256, M_ANCH, M_PAD, wqkvt, 256, 768, P[31], qkvb, 768, nullptr, 0, 0);
  global_attn_kernel<<<(M_ANCH * NHEAD + 7) / 8, 256, 0, stream>>>(qkvb, og16);
  gemm(og16, 256, M_ANCH, M_PAD, gawot, 256, 256, P[33], sGA, 256, nullptr, 0, 0);
  lnk(sGA, 256, xg0, 256, P[34], P[35], M_ANCH, 0, xg1f, 256, xg1h, 256);
  gemm(xg1h, 256, M_ANCH, M_PAD, gafw1t, 256, 1024, P[39], nullptr, 0, h1g, 1024, 1);
  gemm(h1g, 1024, M_ANCH, M_PAD, gafw2t, 1024, 256, P[41], sGA, 256, nullptr, 0, 0);
  lnk(sGA, 256, xg1f, 256, P[36], P[37], M_ANCH, 0, xg2f, 256, nullptr, 0);

  // --- fusion + final projections ---
  upsample_kernel<<<N_PTS, 256, 0, stream>>>(xg2f, cat1, cat2);
  gemm(cat1, 512, N_PTS, N_PTS, fuwt, 512, 256, P[43], sA, 256, nullptr, 0, 0);
  lnk(sA, 256, nullptr, 0, P[44], P[45], N_PTS, 1, nullptr, 0, cat2 + 256, 512);
  gemm(cat2, 512, N_PTS, N_PTS, fpw1t, 512, 256, P[47], sA, 256, nullptr, 0, 0);
  lnk(sA, 256, nullptr, 0, P[48], P[49], N_PTS, 1, nullptr, 0, h1f, 256);
  gemm(h1f, 256, N_PTS, N_PTS, fpw2t, 256, 256, P[51], sA, 256, nullptr, 0, 0);
  lnk(sA, 256, nullptr, 0, P[52], P[53], N_PTS, 1, out, 256, nullptr, 0);
}
